// SparseMoE_8461085573277
// MI455X (gfx1250) — matrix operand order repaired and checked by reading, not compiled, not run on hardware
//
#include <hip/hip_runtime.h>

#define N_TOK 16384
#define D_IN  1024
#define H_DIM 4096
#define E_NUM 8
#define TM 64
#define TNB 128
#define TK 64

typedef __attribute__((ext_vector_type(16))) __bf16 v16bf;
typedef __attribute__((ext_vector_type(8)))  __bf16 v8bf;
typedef __attribute__((ext_vector_type(8)))  float  v8f;
typedef __attribute__((ext_vector_type(4)))  int    v4i;
typedef __attribute__((address_space(1))) v4i v4i_g;   // global
typedef __attribute__((address_space(3))) v4i v4i_l;   // LDS

__device__ __forceinline__ __bf16 f2bf(float f) {
  unsigned u = __builtin_bit_cast(unsigned, f);
  u += 0x7FFFu + ((u >> 16) & 1u);           // round-to-nearest-even
  unsigned short s = (unsigned short)(u >> 16);
  return __builtin_bit_cast(__bf16, s);
}

__device__ __forceinline__ float gelu_tanh(float v) {
  float t = 0.7978845608028654f * (v + 0.044715f * v * v * v);
  return 0.5f * v * (1.0f + tanhf(t));
}

// ---------------- async global->LDS copy (CDNA5 ASYNCcnt path) ----------------
#if __has_builtin(__builtin_amdgcn_global_load_async_to_lds_b128)
#define ASYNC_CP 1
#endif

__device__ __forceinline__ void cp_b128_to_lds(const __bf16* g, __bf16* l) {
#ifdef ASYNC_CP
  v4i_g* gp = (v4i_g*)(unsigned long long)(size_t)g;
  v4i_l* lp = (v4i_l*)(unsigned int)(size_t)l;
  __builtin_amdgcn_global_load_async_to_lds_b128(gp, lp, 0, 0);
#else
  *(v8bf*)l = *(const v8bf*)g;
#endif
}

__device__ __forceinline__ void cp_async_wait() {
#ifdef ASYNC_CP
#if __has_builtin(__builtin_amdgcn_s_wait_asynccnt)
  __builtin_amdgcn_s_wait_asynccnt(0);
#else
  asm volatile("s_wait_asynccnt 0x0" ::: "memory");
#endif
  asm volatile("" ::: "memory");
#endif
}

// ---------------- zero + convert ----------------
__global__ void zero_kernel(float* __restrict__ out, long n, int* __restrict__ counts) {
  long i = (long)blockIdx.x * blockDim.x + threadIdx.x;
  long stride = (long)gridDim.x * blockDim.x;
  for (; i < n; i += stride) out[i] = 0.0f;
  if (blockIdx.x == 0 && threadIdx.x < E_NUM) counts[threadIdx.x] = 0;
}

__global__ void convert_kernel(const float* __restrict__ src, __bf16* __restrict__ dst, long n) {
  long i = (long)blockIdx.x * blockDim.x + threadIdx.x;
  long stride = (long)gridDim.x * blockDim.x;
  for (; i < n; i += stride) dst[i] = f2bf(src[i]);
}

// Tiled transpose+convert: src [K][N] f32 (per expert) -> dst [N][K] bf16 (per expert).
// Coalesced reads and writes via a 32x32 LDS tile. Grid: (K/32, N/32, E), 256 threads.
__global__ __launch_bounds__(256)
void transpose_convert_kernel(const float* __restrict__ src, __bf16* __restrict__ dst,
                              int K, int N) {
  __shared__ float tile[32][33];
  long eoff = (long)blockIdx.z * K * N;
  int kb = blockIdx.x * 32, nb = blockIdx.y * 32;
  int tx = threadIdx.x & 31, ty = threadIdx.x >> 5;
#pragma unroll
  for (int i = 0; i < 4; i++) {
    int r = ty + i * 8;
    tile[r][tx] = src[eoff + (long)(kb + r) * N + nb + tx];
  }
  __syncthreads();
#pragma unroll
  for (int i = 0; i < 4; i++) {
    int r = ty + i * 8;
    dst[eoff + (long)(nb + r) * K + kb + tx] = f2bf(tile[tx][r]);
  }
}

// ---------------- router ----------------
__device__ __forceinline__ void accum_seg(const float* __restrict__ f, int len,
                                          const float* __restrict__ Wr, float* acc) {
  for (int i = 0; i < len; i++) {
    float v = f[i];
#pragma unroll
    for (int e = 0; e < E_NUM; e++) acc[e] += v * Wr[i * E_NUM + e];
  }
}

__global__ __launch_bounds__(256)
void router_kernel(const float* __restrict__ x, const int* __restrict__ city_idx_p,
                   const float* __restrict__ dt, const float* __restrict__ dd,
                   const float* __restrict__ drg, const float* __restrict__ dent,
                   const float* __restrict__ cemb, const float* __restrict__ Wr,
                   const float* __restrict__ br, float* __restrict__ gate1_out,
                   float* __restrict__ gating, int* __restrict__ counts,
                   int* __restrict__ lists) {
  int n = blockIdx.x * blockDim.x + threadIdx.x;
  if (n >= N_TOK) return;
  int ci = city_idx_p[0];
  float acc[E_NUM];
#pragma unroll
  for (int e = 0; e < E_NUM; e++) acc[e] = br[e];
  long j = 0;
  accum_seg(x + (long)n * 1024, 1024, Wr + j * E_NUM, acc); j += 1024;
  accum_seg(cemb + ci * 32,       32, Wr + j * E_NUM, acc); j += 32;
  accum_seg(dt  + (long)n * 256, 256, Wr + j * E_NUM, acc); j += 256;
  accum_seg(dd  + (long)n * 256, 256, Wr + j * E_NUM, acc); j += 256;
  accum_seg(drg + (long)n * 128, 128, Wr + j * E_NUM, acc); j += 128;
  accum_seg(dent+ (long)n * 128, 128, Wr + j * E_NUM, acc); j += 128;

  // full softmax -> gate1 output
  float m = acc[0];
#pragma unroll
  for (int e = 1; e < E_NUM; e++) m = fmaxf(m, acc[e]);
  float s = 0.0f, ex[E_NUM];
#pragma unroll
  for (int e = 0; e < E_NUM; e++) { ex[e] = expf(acc[e] - m); s += ex[e]; }
  float inv = 1.0f / s;
#pragma unroll
  for (int e = 0; e < E_NUM; e++) gate1_out[(long)n * E_NUM + e] = ex[e] * inv;

  // top-2 (first-index wins on ties, matching lax.top_k)
  int i0 = 0;
#pragma unroll
  for (int e = 1; e < E_NUM; e++) if (acc[e] > acc[i0]) i0 = e;
  int i1 = (i0 == 0) ? 1 : 0;
#pragma unroll
  for (int e = 0; e < E_NUM; e++) if (e != i0 && acc[e] > acc[i1]) i1 = e;

  float mm = fmaxf(acc[i0], acc[i1]);
  float g0 = expf(acc[i0] - mm), g1 = expf(acc[i1] - mm);
  float gi = 1.0f / (g0 + g1);
  g0 *= gi; g1 *= gi;
#pragma unroll
  for (int e = 0; e < E_NUM; e++) gating[(long)n * E_NUM + e] = 0.0f;
  gating[(long)n * E_NUM + i0] = g0;
  gating[(long)n * E_NUM + i1] = g1;

  int s0 = atomicAdd(&counts[i0], 1); lists[i0 * N_TOK + s0] = n;
  int s1 = atomicAdd(&counts[i1], 1); lists[i1 * N_TOK + s1] = n;
}

// ---------------- WMMA fragment helpers ----------------
__device__ __forceinline__ v16bf load_a_frag(const __bf16* Arow /* &As[row][kk] */, int lane) {
  int kb = (lane < 16) ? 0 : 16;
  v8bf lo = *(const v8bf*)(Arow + kb);
  v8bf hi = *(const v8bf*)(Arow + kb + 8);
  return __builtin_shufflevector(lo, hi, 0,1,2,3,4,5,6,7,8,9,10,11,12,13,14,15);
}

// ---------------- GEMM 1: h = gelu(Xg @ Wfc[e] + bfc[e]) -> hact (bf16) ----------------
// Block tile 64x128, K-chunk 64. Wave tile M16 x N64: 8 WMMAs per chunk.
// A (row-major) and B (pre-transposed [N][K]) both staged via async global->LDS b128.
__global__ __launch_bounds__(256)
void gemm_fc_kernel(const __bf16* __restrict__ xbf, const __bf16* __restrict__ wfcT,
                    const float* __restrict__ bfc, const int* __restrict__ counts,
                    const int* __restrict__ lists, __bf16* __restrict__ hact, int e) {
  __shared__ __bf16 As[TM][TK];
  __shared__ __bf16 Bt[TNB][TK];
  int cnt = counts[e];
  int m0 = blockIdx.x * TM;
  if (m0 >= cnt) return;
  int n0 = blockIdx.y * TNB;
  int t = threadIdx.x, lane = t & 31, wid = t >> 5;
  int wm = wid & 3, wn = wid >> 2;   // wave tile: rows [wm*16,+16), cols [wn*64,+64)

  const __bf16* wfe = wfcT + (long)e * D_IN * H_DIM;   // [H_DIM][D_IN]
  const int* lst = lists + e * N_TOK;

  // A staging: 64x64 tile, 256 threads x 32B (2 async b128)
  int ar = t >> 2, aseg = t & 3;
  int aslot = m0 + ar;
  int atok = lst[(aslot < cnt) ? aslot : 0];
  const __bf16* asrc = xbf + (long)atok * D_IN + aseg * 16;

  // B staging: 128x64 tile, 256 threads x 64B (4 async b128)
  int brow = t >> 1, bseg = t & 1;
  const __bf16* bsrc = wfe + (long)(n0 + brow) * D_IN + bseg * 32;

  v8f c[4] = {{}, {}, {}, {}};
  for (int k0 = 0; k0 < D_IN; k0 += TK) {
    __syncthreads();
    cp_b128_to_lds(asrc + k0,     &As[ar][aseg * 16]);
    cp_b128_to_lds(asrc + k0 + 8, &As[ar][aseg * 16 + 8]);
#pragma unroll
    for (int q = 0; q < 4; q++)
      cp_b128_to_lds(bsrc + k0 + q * 8, &Bt[brow][bseg * 32 + q * 8]);
    cp_async_wait();
    __syncthreads();

#pragma unroll
    for (int kk = 0; kk < TK; kk += 32) {
      v16bf a = load_a_frag(&As[wm * 16 + (lane & 15)][kk], lane);
      int kh = kk + ((lane < 16) ? 0 : 16);
#pragma unroll
      for (int f = 0; f < 4; f++) {
        v16bf b = *(const v16bf*)&Bt[wn * 64 + f * 16 + (lane & 15)][kh];
        c[f] = __builtin_amdgcn_wmma_f32_16x16x32_bf16(false, a, false, b, (short)0, c[f],
                                                       false, false);
      }
    }
  }

  int nloc = lane & 15;
  int mbase = m0 + wm * 16 + ((lane < 16) ? 0 : 8);
#pragma unroll
  for (int r = 0; r < 8; r++) {
    int slot = mbase + r;
    if (slot < cnt) {
#pragma unroll
      for (int f = 0; f < 4; f++) {
        int h = n0 + wn * 64 + f * 16 + nloc;
        float v = gelu_tanh(c[f][r] + bfc[e * H_DIM + h]);
        hact[(long)slot * H_DIM + h] = f2bf(v);
      }
    }
  }
}

// ---------------- GEMM 2: out[token] += g * (hact @ Wproj[e] + bproj[e]) ----------------
__global__ __launch_bounds__(256)
void gemm_proj_kernel(const __bf16* __restrict__ hact, const __bf16* __restrict__ wprojT,
                      const float* __restrict__ bproj, const int* __restrict__ counts,
                      const int* __restrict__ lists, const float* __restrict__ gating,
                      float* __restrict__ out, int e) {
  __shared__ __bf16 As[TM][TK];
  __shared__ __bf16 Bt[TNB][TK];
  int cnt = counts[e];
  int m0 = blockIdx.x * TM;
  if (m0 >= cnt) return;
  int n0 = blockIdx.y * TNB;
  int t = threadIdx.x, lane = t & 31, wid = t >> 5;
  int wm = wid & 3, wn = wid >> 2;

  const __bf16* wpe = wprojT + (long)e * H_DIM * D_IN;  // [D_IN][H_DIM]
  const int* lst = lists + e * N_TOK;

  int ar = t >> 2, aseg = t & 3;
  const __bf16* asrc = hact + (long)(m0 + ar) * H_DIM + aseg * 16;
  int brow = t >> 1, bseg = t & 1;
  const __bf16* bsrc = wpe + (long)(n0 + brow) * H_DIM + bseg * 32;

  v8f c[4] = {{}, {}, {}, {}};
  for (int k0 = 0; k0 < H_DIM; k0 += TK) {
    __syncthreads();
    cp_b128_to_lds(asrc + k0,     &As[ar][aseg * 16]);
    cp_b128_to_lds(asrc + k0 + 8, &As[ar][aseg * 16 + 8]);
#pragma unroll
    for (int q = 0; q < 4; q++)
      cp_b128_to_lds(bsrc + k0 + q * 8, &Bt[brow][bseg * 32 + q * 8]);
    cp_async_wait();
    __syncthreads();

#pragma unroll
    for (int kk = 0; kk < TK; kk += 32) {
      v16bf a = load_a_frag(&As[wm * 16 + (lane & 15)][kk], lane);
      int kh = kk + ((lane < 16) ? 0 : 16);
#pragma unroll
      for (int f = 0; f < 4; f++) {
        v16bf b = *(const v16bf*)&Bt[wn * 64 + f * 16 + (lane & 15)][kh];
        c[f] = __builtin_amdgcn_wmma_f32_16x16x32_bf16(false, a, false, b, (short)0, c[f],
                                                       false, false);
      }
    }
  }

  int nloc = lane & 15;
  int mbase = wm * 16 + ((lane < 16) ? 0 : 8);
#pragma unroll
  for (int r = 0; r < 8; r++) {
    int slot = m0 + mbase + r;
    if (slot < cnt) {
      int token = lst[slot];
      float g = gating[(long)token * E_NUM + e];
      float* op = out + (long)token * D_IN;
#pragma unroll
      for (int f = 0; f < 4; f++) {
        int col = n0 + wn * 64 + f * 16 + nloc;
        op[col] += g * (c[f][r] + bproj[e * D_IN + col]);
      }
    }
  }
}

// ---------------- host ----------------
static inline void* ws_take(char*& p, size_t bytes) {
  void* r = (void*)p;
  p += (bytes + 255) & ~(size_t)255;
  return r;
}

extern "C" void kernel_launch(void* const* d_in, const int* in_sizes, int n_in,
                              void* d_out, int out_size, void* d_ws, size_t ws_size,
                              hipStream_t stream) {
  const float* x     = (const float*)d_in[0];
  const int*   city  = (const int*)d_in[1];
  const float* dt    = (const float*)d_in[2];
  const float* dd    = (const float*)d_in[3];
  const float* drg   = (const float*)d_in[4];
  const float* dent  = (const float*)d_in[5];
  const float* cemb  = (const float*)d_in[6];
  const float* Wr    = (const float*)d_in[7];
  const float* br    = (const float*)d_in[8];
  const float* Wfc   = (const float*)d_in[9];
  const float* bfc   = (const float*)d_in[10];
  const float* Wproj = (const float*)d_in[11];
  const float* bproj = (const float*)d_in[12];

  float* out   = (float*)d_out;                       // [N, D]
  float* gate1 = out + (long)N_TOK * D_IN;            // [N, E]

  char* p = (char*)d_ws;
  int*    counts  = (int*)   ws_take(p, E_NUM * sizeof(int));
  float*  gating  = (float*) ws_take(p, (size_t)N_TOK * E_NUM * sizeof(float));
  int*    lists   = (int*)   ws_take(p, (size_t)E_NUM * N_TOK * sizeof(int));
  __bf16* xbf     = (__bf16*)ws_take(p, (size_t)N_TOK * D_IN * sizeof(__bf16));
  __bf16* wfcT    = (__bf16*)ws_take(p, (size_t)E_NUM * D_IN * H_DIM * sizeof(__bf16));
  __bf16* wprojT  = (__bf16*)ws_take(p, (size_t)E_NUM * H_DIM * D_IN * sizeof(__bf16));
  __bf16* hact    = (__bf16*)ws_take(p, (size_t)N_TOK * H_DIM * sizeof(__bf16));

  zero_kernel<<<2048, 256, 0, stream>>>(out, (long)N_TOK * D_IN, counts);
  convert_kernel<<<4096, 256, 0, stream>>>(x, xbf, (long)N_TOK * D_IN);
  // Wfc [e][K=D_IN][N=H_DIM] -> wfcT [e][N=H_DIM][K=D_IN]
  transpose_convert_kernel<<<dim3(D_IN / 32, H_DIM / 32, E_NUM), 256, 0, stream>>>(
      Wfc, wfcT, D_IN, H_DIM);
  // Wproj [e][K=H_DIM][N=D_IN] -> wprojT [e][N=D_IN][K=H_DIM]
  transpose_convert_kernel<<<dim3(H_DIM / 32, D_IN / 32, E_NUM), 256, 0, stream>>>(
      Wproj, wprojT, H_DIM, D_IN);
  router_kernel<<<N_TOK / 256, 256, 0, stream>>>(x, city, dt, dd, drg, dent, cemb, Wr, br,
                                                 gate1, gating, counts, lists);
  for (int e = 0; e < E_NUM; e++) {
    gemm_fc_kernel<<<dim3(N_TOK / TM, H_DIM / TNB), 256, 0, stream>>>(
        xbf, wfcT, bfc, counts, lists, hact, e);
    gemm_proj_kernel<<<dim3(N_TOK / TM, D_IN / TNB), 256, 0, stream>>>(
        hact, wprojT, bproj, counts, lists, gating, out, e);
  }
}
